// LSTMPINN_86088324481437
// MI455X (gfx1250) — matrix operand order repaired and checked by reading, not compiled, not run on hardware
//
#include <hip/hip_runtime.h>

typedef __bf16 bf16_t;
typedef __attribute__((ext_vector_type(16))) __bf16 v16bf;
typedef __attribute__((ext_vector_type(8)))  __bf16 v8bf;
typedef __attribute__((ext_vector_type(8)))  float  v8f;

#define HDIM   128
#define FCDIM  256
#define NPTS   65536
#define WAVES  2
#define TSTEPS 6

// workspace offsets (bf16 elements)
#define WS_WHH0 0
#define WS_WIH1 65536
#define WS_WHH1 131072
#define WS_WD0  196608
#define WS_WD1  262144
#define WS_WD2  327680
#define WS_TOTAL 393216

// LDS layout (bytes): 5 regions x 8192 B per wave = 80 KB per 2-wave block
#define LDS_H0   0
#define LDS_H1   (LDS_H0 + WAVES * 8192)
#define LDS_C0   (LDS_H1 + WAVES * 8192)
#define LDS_C1   (LDS_C0 + WAVES * 8192)
#define LDS_HSUM (LDS_C1 + WAVES * 8192)
#define LDS_TOTAL (LDS_HSUM + WAVES * 8192)

// sched_group_barrier masks
#define SG_WMMA 0x008
#define SG_VMEM_READ 0x020

// fast transcendentals: v_exp_f32 / v_rcp_f32 (TRANS ops co-execute with VALU/XDL)
__device__ __forceinline__ float sigm(float v) {
  return __builtin_amdgcn_rcpf(1.0f + __expf(-v));
}
__device__ __forceinline__ float tanh_fast(float x) {
  float e = __expf(2.0f * x);
  return 1.0f - 2.0f * __builtin_amdgcn_rcpf(e + 1.0f);
}

// A-fragment (16x32 bf16, MxK) from LDS row-major h[16][K]
__device__ __forceinline__ v16bf load_a_frag(const bf16_t* h, int K, int kbase, int lane) {
  const int m = lane & 15, half = lane >> 4;
  const bf16_t* p = h + m * K + kbase + half * 8;
  v8bf lo = *(const v8bf*)p;
  v8bf hi = *(const v8bf*)(p + 16);
  v16bf a;
#pragma unroll
  for (int i = 0; i < 8; ++i) { a[i] = lo[i]; a[i + 8] = hi[i]; }
  return a;
}

// B-fragment (32x16 bf16, KxN) from global row-major W[outdim][K]
__device__ __forceinline__ v16bf load_b_frag(const bf16_t* W, int K, int nbase, int kbase, int lane) {
  const int n = nbase + (lane & 15), half = lane >> 4;
  const bf16_t* p = W + n * K + kbase + half * 8;
  v8bf lo = *(const v8bf*)p;
  v8bf hi = *(const v8bf*)(p + 16);
  v16bf b;
#pragma unroll
  for (int i = 0; i < 8; ++i) { b[i] = lo[i]; b[i + 8] = hi[i]; }
  return b;
}

__device__ __forceinline__ v8f wmma_bf16(v16bf a, v16bf b, v8f c) {
  return __builtin_amdgcn_wmma_f32_16x16x32_bf16(false, a, false, b, (short)0, c, false, false);
}

// one gate-block GEMM step: batch 32 global_load_b128, then 16 WMMAs.
__device__ __forceinline__ void gemm_gates(const v16bf* af, const bf16_t* W,
                                           int jj, int lane,
                                           v8f& gi, v8f& gf_, v8f& gg, v8f& go) {
  v16bf bf_[16];
#pragma unroll
  for (int k = 0; k < 4; ++k) {
#pragma unroll
    for (int g = 0; g < 4; ++g)
      bf_[k * 4 + g] = load_b_frag(W, HDIM, g * HDIM + jj * 16, k * 32, lane);
  }
#pragma unroll
  for (int k = 0; k < 4; ++k) {
    gi  = wmma_bf16(af[k], bf_[k * 4 + 0], gi);
    gf_ = wmma_bf16(af[k], bf_[k * 4 + 1], gf_);
    gg  = wmma_bf16(af[k], bf_[k * 4 + 2], gg);
    go  = wmma_bf16(af[k], bf_[k * 4 + 3], go);
  }
  __builtin_amdgcn_sched_group_barrier(SG_VMEM_READ, 32, 0);
  __builtin_amdgcn_sched_group_barrier(SG_WMMA, 16, 0);
}

// fused double GEMM block for layer 1 (input W1 + recurrent W2).
// Scheduling: bA burst (32 VMEM) -> 8 x [4 VMEM (bB) : 2 WMMA (A-set)] -> 16 WMMA (B-set).
// bA fragments die as bB arrives, keeping peak B pressure ~128 VGPRs (no spills)
// while always holding >=32 loads in flight.
__device__ __forceinline__ void gemm_gates_dual(const v16bf* a1f, const bf16_t* W1,
                                                const v16bf* a2f, const bf16_t* W2,
                                                int jj, int lane,
                                                v8f& gi, v8f& gf_, v8f& gg, v8f& go) {
  v16bf bA[16], bB[16];
#pragma unroll
  for (int k = 0; k < 4; ++k) {
#pragma unroll
    for (int g = 0; g < 4; ++g)
      bA[k * 4 + g] = load_b_frag(W1, HDIM, g * HDIM + jj * 16, k * 32, lane);
  }
#pragma unroll
  for (int k = 0; k < 4; ++k) {
#pragma unroll
    for (int g = 0; g < 4; ++g)
      bB[k * 4 + g] = load_b_frag(W2, HDIM, g * HDIM + jj * 16, k * 32, lane);
  }
#pragma unroll
  for (int k = 0; k < 4; ++k) {
    gi  = wmma_bf16(a1f[k], bA[k * 4 + 0], gi);
    gf_ = wmma_bf16(a1f[k], bA[k * 4 + 1], gf_);
    gg  = wmma_bf16(a1f[k], bA[k * 4 + 2], gg);
    go  = wmma_bf16(a1f[k], bA[k * 4 + 3], go);
  }
#pragma unroll
  for (int k = 0; k < 4; ++k) {
    gi  = wmma_bf16(a2f[k], bB[k * 4 + 0], gi);
    gf_ = wmma_bf16(a2f[k], bB[k * 4 + 1], gf_);
    gg  = wmma_bf16(a2f[k], bB[k * 4 + 2], gg);
    go  = wmma_bf16(a2f[k], bB[k * 4 + 3], go);
  }
  __builtin_amdgcn_sched_group_barrier(SG_VMEM_READ, 32, 0);
#pragma unroll
  for (int i = 0; i < 8; ++i) {
    __builtin_amdgcn_sched_group_barrier(SG_VMEM_READ, 4, 0);
    __builtin_amdgcn_sched_group_barrier(SG_WMMA, 2, 0);
  }
  __builtin_amdgcn_sched_group_barrier(SG_WMMA, 16, 0);
}

// f32 -> bf16 (round to nearest even) weight repack into workspace
__global__ void convert_weights(const float* __restrict__ a0, const float* __restrict__ a1,
                                const float* __restrict__ a2, const float* __restrict__ a3,
                                const float* __restrict__ a4, const float* __restrict__ a5,
                                unsigned short* __restrict__ ws) {
  int idx = blockIdx.x * blockDim.x + threadIdx.x;
  if (idx >= WS_TOTAL) return;
  int which = idx >> 16;
  int off = idx & 65535;
  const float* src = (which == 0) ? a0 : (which == 1) ? a1 : (which == 2) ? a2
                   : (which == 3) ? a3 : (which == 4) ? a4 : a5;
  union { float f; unsigned int u; } v;
  v.f = src[off];
  unsigned int r = v.u + 0x7FFFu + ((v.u >> 16) & 1u);
  ws[idx] = (unsigned short)(r >> 16);
}

__global__ void __launch_bounds__(WAVES * 32) __attribute__((amdgpu_waves_per_eu(2)))
lstm_pinn_fused(const float* __restrict__ gx, const float* __restrict__ gy,
                const float* __restrict__ Wih0, const float* __restrict__ b0,
                const float* __restrict__ b1,
                const float* __restrict__ bd0, const float* __restrict__ bd1,
                const float* __restrict__ bd2,
                const float* __restrict__ Wout, const float* __restrict__ bout,
                const bf16_t* __restrict__ ws, float* __restrict__ out) {
  extern __shared__ char smem[];
  const int lane = threadIdx.x & 31;
  const int w    = threadIdx.x >> 5;
  const int nl   = lane & 15;
  const int half = lane >> 4;
  const int pbase = blockIdx.x * (WAVES * 16) + w * 16;

  bf16_t* h0W   = (bf16_t*)(smem + LDS_H0)   + w * 4096;  // 2 x 16x128 bf16
  bf16_t* h1W   = (bf16_t*)(smem + LDS_H1)   + w * 4096;  // 2 x 16x128 bf16
  float*  c0W   = (float*) (smem + LDS_C0)   + w * 2048;  // 16x128 f32
  float*  c1W   = (float*) (smem + LDS_C1)   + w * 2048;  // 16x128 f32
  float*  hsumW = (float*) (smem + LDS_HSUM) + w * 2048;  // 16x128 f32
  bf16_t* featA = h0W;                                    // alias (16x256 bf16)
  bf16_t* featB = h1W;                                    // alias (16x256 bf16)

  const bf16_t* Whh0 = ws + WS_WHH0;
  const bf16_t* Wih1 = ws + WS_WIH1;
  const bf16_t* Whh1 = ws + WS_WHH1;

  // per-lane copies of x,y for the 8 rows this lane owns in the C layout
  float xr[8], yr[8];
#pragma unroll
  for (int r = 0; r < 8; ++r) {
    int m = r + half * 8;
    xr[r] = 2.0f * gx[pbase + m] - 1.0f;
    yr[r] = 2.0f * gy[pbase + m] - 1.0f;
  }

  for (int t = 0; t < TSTEPS; ++t) {
    const bf16_t* h0prev = h0W + ((t + 1) & 1) * 2048;
    bf16_t*       h0cur  = h0W + (t & 1) * 2048;
    const bf16_t* h1prev = h1W + ((t + 1) & 1) * 2048;
    bf16_t*       h1cur  = h1W + (t & 1) * 2048;

    // per-timestep input feature, invariant across jj: hoist out of the hot loop
    float sv[8];
#pragma unroll
    for (int r = 0; r < 8; ++r) {
      float xv = xr[r], yv = yr[r];
      sv[r] = (t == 0) ? xv
            : (t == 1) ? yv
            : (t == 2) ? xv + yv
            : (t == 3) ? xv - yv
            : (t == 4) ? xv * yv
            :            xv * xv + yv * yv;
    }

    // ---------------- LSTM layer 0 ----------------
    v16bf a0f[4];
    if (t > 0) {
#pragma unroll
      for (int k = 0; k < 4; ++k) a0f[k] = load_a_frag(h0prev, HDIM, k * 32, lane);
    }
#pragma unroll
    for (int jj = 0; jj < 8; ++jj) {
      const int nglob = jj * 16 + nl;
      v8f gi, gf_, gg, go;
      {
        float bi = b0[nglob], bf = b0[HDIM + nglob];
        float bg = b0[2 * HDIM + nglob], bo = b0[3 * HDIM + nglob];
#pragma unroll
        for (int r = 0; r < 8; ++r) { gi[r] = bi; gf_[r] = bf; gg[r] = bg; go[r] = bo; }
      }
      if (t > 0)  // h_{t-1} contribution: [16x128] x [128x512]
        gemm_gates(a0f, Whh0, jj, lane, gi, gf_, gg, go);
      // scalar input (input size = 1): gates += s_t[m] * W_ih_l0[n][0]
      float wi = Wih0[nglob], wf = Wih0[HDIM + nglob];
      float wg = Wih0[2 * HDIM + nglob], wo = Wih0[3 * HDIM + nglob];
#pragma unroll
      for (int r = 0; r < 8; ++r) {
        float s = sv[r];
        float vi = gi[r] + s * wi, vf = gf_[r] + s * wf;
        float vg = gg[r] + s * wg, vo = go[r] + s * wo;
        const int idx = (r + half * 8) * HDIM + nglob;
        float cv = sigm(vi) * tanh_fast(vg);
        if (t > 0) cv += sigm(vf) * c0W[idx];
        c0W[idx] = cv;
        float hv = sigm(vo) * tanh_fast(cv);
        h0cur[idx] = (bf16_t)hv;
      }
    }
    __syncthreads();

    // ---------------- LSTM layer 1 ----------------
    v16bf aIn[4], aH[4];
#pragma unroll
    for (int k = 0; k < 4; ++k) aIn[k] = load_a_frag(h0cur, HDIM, k * 32, lane);
    if (t > 0) {
#pragma unroll
      for (int k = 0; k < 4; ++k) aH[k] = load_a_frag(h1prev, HDIM, k * 32, lane);
    }
#pragma unroll
    for (int jj = 0; jj < 8; ++jj) {
      const int nglob = jj * 16 + nl;
      v8f gi, gf_, gg, go;
      {
        float bi = b1[nglob], bf = b1[HDIM + nglob];
        float bg = b1[2 * HDIM + nglob], bo = b1[3 * HDIM + nglob];
#pragma unroll
        for (int r = 0; r < 8; ++r) { gi[r] = bi; gf_[r] = bf; gg[r] = bg; go[r] = bo; }
      }
      if (t > 0)
        gemm_gates_dual(aIn, Wih1, aH, Whh1, jj, lane, gi, gf_, gg, go);
      else
        gemm_gates(aIn, Wih1, jj, lane, gi, gf_, gg, go);
#pragma unroll
      for (int r = 0; r < 8; ++r) {
        const int idx = (r + half * 8) * HDIM + nglob;
        float cv = sigm(gi[r]) * tanh_fast(gg[r]);
        if (t > 0) cv += sigm(gf_[r]) * c1W[idx];
        c1W[idx] = cv;
        float hv = sigm(go[r]) * tanh_fast(cv);
        h1cur[idx] = (bf16_t)hv;
        if (t == 0) hsumW[idx] = hv;
        else        hsumW[idx] = hsumW[idx] + hv;
      }
    }
    __syncthreads();
  }

  // ---------------- feature build: [last_h1 | mean_h1] (16x256 bf16) ----------------
  const bf16_t* h1fin = h1W + ((TSTEPS - 1) & 1) * 2048;
  for (int i = lane; i < 256; i += 32) {         // 16 rows x 16 chunks of 8 bf16
    int m = i >> 4, ch = i & 15;
    *(v8bf*)(featA + m * FCDIM + ch * 8) = *(const v8bf*)(h1fin + m * HDIM + ch * 8);
  }
#pragma unroll
  for (int jj = 0; jj < 8; ++jj) {
#pragma unroll
    for (int r = 0; r < 8; ++r) {
      int m = r + half * 8;
      featA[m * FCDIM + HDIM + jj * 16 + nl] =
          (bf16_t)(hsumW[m * HDIM + jj * 16 + nl] * (1.0f / 6.0f));
    }
  }
  __syncthreads();

  // ---------------- dense head: 3 x (tanh([16x256] x [256x256] + b)) ----------------
#pragma unroll
  for (int d = 0; d < 3; ++d) {
    const bf16_t* Wd = (d == 0) ? (ws + WS_WD0) : (d == 1) ? (ws + WS_WD1) : (ws + WS_WD2);
    const float*  bb = (d == 0) ? bd0 : (d == 1) ? bd1 : bd2;
    const bf16_t* A  = (d & 1) ? featB : featA;
    bf16_t*       O  = (d & 1) ? featA : featB;
    v16bf af[8];
#pragma unroll
    for (int k = 0; k < 8; ++k) af[k] = load_a_frag(A, FCDIM, k * 32, lane);
    for (int nt = 0; nt < 16; ++nt) {
      v8f acc;
      float bv = bb[nt * 16 + nl];
#pragma unroll
      for (int r = 0; r < 8; ++r) acc[r] = bv;
      v16bf bfr[8];
#pragma unroll
      for (int k = 0; k < 8; ++k)
        bfr[k] = load_b_frag(Wd, FCDIM, nt * 16, k * 32, lane);
#pragma unroll
      for (int k = 0; k < 8; ++k)
        acc = wmma_bf16(af[k], bfr[k], acc);
      __builtin_amdgcn_sched_group_barrier(SG_VMEM_READ, 16, 0);
      __builtin_amdgcn_sched_group_barrier(SG_WMMA, 8, 0);
#pragma unroll
      for (int r = 0; r < 8; ++r)
        O[(r + half * 8) * FCDIM + nt * 16 + nl] = (bf16_t)tanh_fast(acc[r]);
    }
    __syncthreads();
  }

  // ---------------- output: [16x256] x [256x4] + bout ----------------
  const bf16_t* F = featB;  // d=2 wrote featB
  for (int p = lane; p < 64; p += 32) {
    int m = p >> 2, o = p & 3;
    float s = bout[o];
#pragma unroll 8
    for (int k = 0; k < FCDIM; ++k)
      s += (float)F[m * FCDIM + k] * Wout[o * FCDIM + k];
    out[(pbase + m) * 4 + o] = s;
  }
}

extern "C" void kernel_launch(void* const* d_in, const int* in_sizes, int n_in,
                              void* d_out, int out_size, void* d_ws, size_t ws_size,
                              hipStream_t stream) {
  const float* x     = (const float*)d_in[0];
  const float* y     = (const float*)d_in[1];
  const float* Wih0  = (const float*)d_in[2];
  const float* Whh0  = (const float*)d_in[3];
  const float* b0    = (const float*)d_in[4];
  const float* Wih1  = (const float*)d_in[5];
  const float* Whh1  = (const float*)d_in[6];
  const float* b1    = (const float*)d_in[7];
  const float* Wd0   = (const float*)d_in[8];
  const float* bd0   = (const float*)d_in[9];
  const float* Wd1   = (const float*)d_in[10];
  const float* bd1   = (const float*)d_in[11];
  const float* Wd2   = (const float*)d_in[12];
  const float* bd2   = (const float*)d_in[13];
  const float* Wout  = (const float*)d_in[14];
  const float* bout  = (const float*)d_in[15];

  unsigned short* ws16 = (unsigned short*)d_ws;
  convert_weights<<<(WS_TOTAL + 255) / 256, 256, 0, stream>>>(
      Whh0, Wih1, Whh1, Wd0, Wd1, Wd2, ws16);

  const int blocks = NPTS / (WAVES * 16);  // 2048
  lstm_pinn_fused<<<blocks, WAVES * 32, LDS_TOTAL, stream>>>(
      x, y, Wih0, b0, b1, bd0, bd1, bd2, Wout, bout,
      (const bf16_t*)d_ws, (float*)d_out);
}
